// GCNNet_43130061586774
// MI455X (gfx1250) — matrix operand order repaired and checked by reading, not compiled, not run on hardware
//
#include <hip/hip_runtime.h>
#include <cstdint>
#include <cstddef>

// ---------------------------------------------------------------------------
// GCNNet on MI455X (gfx1250, wave32).
//  - Matmuls: v_wmma_f32_16x16x32_bf16, fully unrolled K loop (templated Cin),
//    A staged via LDS as packed bf16 pairs (ds_store_b32 / ds_load_b128),
//    W pre-packed into this kernel's B-fragment k order (k = 16*seg + e) (global_load_b128).
//  - Graph aggregation: device-built CSR (histogram+scan+fill), atomic-free
//    per-node gather-reduce; features (51MB) resident in 192MB L2.
//  - BN folded to per-channel scale/shift everywhere.
// ---------------------------------------------------------------------------

#define NN 50000
#define NE 800000
#define NG 500

typedef __bf16 bf16;
typedef bf16  v16bf __attribute__((ext_vector_type(16)));
typedef bf16  v8bf  __attribute__((ext_vector_type(8)));
typedef bf16  v2bf  __attribute__((ext_vector_type(2)));
typedef float v8f   __attribute__((ext_vector_type(8)));

// ------------------------------ utility kernels ----------------------------
__global__ void k_fill_f32(float* __restrict__ p, float v, int n) {
    int i = blockIdx.x * blockDim.x + threadIdx.x;
    if (i < n) p[i] = v;
}
__global__ void k_zero_i32(int* __restrict__ p, int n) {
    int i = blockIdx.x * blockDim.x + threadIdx.x;
    if (i < n) p[i] = 0;
}
__global__ void k_copy_i32(int* __restrict__ d, const int* __restrict__ s, int n) {
    int i = blockIdx.x * blockDim.x + threadIdx.x;
    if (i < n) d[i] = s[i];
}
// deg[i] starts at 1.0 (self-loop); count real edges on dst.
__global__ void k_count_deg(const int* __restrict__ dst, float* __restrict__ deg) {
    int e = blockIdx.x * blockDim.x + threadIdx.x;
    if (e < NE) atomicAdd(&deg[dst[e]], 1.0f);
}
__global__ void k_rsqrt_inplace(float* __restrict__ p, int n) {
    int i = blockIdx.x * blockDim.x + threadIdx.x;
    if (i < n) p[i] = rsqrtf(p[i]);
}
__global__ void k_hist(const int* __restrict__ dst, int* __restrict__ cnts) {
    int e = blockIdx.x * blockDim.x + threadIdx.x;
    if (e < NE) atomicAdd(&cnts[dst[e]], 1);
}
// single-block exclusive scan over n=50000 counts -> rowptr[0..n]
__global__ __launch_bounds__(1024) void k_scan(const int* __restrict__ cnts, int n,
                                               int* __restrict__ rowptr) {
    __shared__ int partial[1024];
    int t = threadIdx.x;
    int chunk = (n + 1023) / 1024;
    int beg = t * chunk;
    int end = beg + chunk; if (end > n) end = n;
    int sum = 0;
    for (int i = beg; i < end; ++i) sum += cnts[i];
    partial[t] = sum;
    __syncthreads();
    if (t == 0) {
        int run = 0;
        for (int i = 0; i < 1024; ++i) { int v = partial[i]; partial[i] = run; run += v; }
        rowptr[n] = run;
    }
    __syncthreads();
    int run = partial[t];
    for (int i = beg; i < end; ++i) { rowptr[i] = run; run += cnts[i]; }
}
// scatter edges into CSR slots; precompute symmetric-norm edge weight
__global__ void k_fill_csr(const int* __restrict__ src, const int* __restrict__ dst,
                           const float* __restrict__ dinv, int* __restrict__ cursor,
                           int* __restrict__ col, float* __restrict__ wgt) {
    int e = blockIdx.x * blockDim.x + threadIdx.x;
    if (e >= NE) return;
    int s = src[e], d = dst[e];
    int pos = atomicAdd(&cursor[d], 1);
    col[pos] = s;
    wgt[pos] = dinv[s] * dinv[d];
}
// fold BN (+ upstream bias) into per-channel scale/shift:
//   y*s + t  ==  ((y + bias) - m) * g/sqrt(v+eps) + b
__global__ void k_bn_prep(const float* __restrict__ g, const float* __restrict__ b,
                          const float* __restrict__ m, const float* __restrict__ v,
                          const float* __restrict__ bias, int C,
                          float* __restrict__ s, float* __restrict__ t) {
    int c = blockIdx.x * blockDim.x + threadIdx.x;
    if (c >= C) return;
    float sc = g[c] * rsqrtf(v[c] + 1e-5f);
    s[c] = sc;
    float bi = bias ? bias[c] : 0.0f;
    t[c] = b[c] - (m[c] - bi) * sc;
}

// Pack W [Cin x Cout] f32 into this kernel's bf16 B-fragment order (contiguous k, matching A):
// Wpk[((ks*2 + seg)*Cout + n)*16 + e] = W[(ks*32 + seg*16 + e)*Cout + n]
// (zero-padded for k >= Cin). Lane (seg = lane>>4, n = col) then reads its
// 16-element fragment as two contiguous 16B global_load_b128.
__global__ void k_pack_w(const float* __restrict__ W, int Cin, int Cout, int ksteps,
                         bf16* __restrict__ Wpk) {
    int i = blockIdx.x * blockDim.x + threadIdx.x;
    int total = ksteps * 2 * Cout * 16;
    if (i >= total) return;
    int e    = i & 15;
    int rest = i >> 4;
    int n    = rest % Cout;
    int sk   = rest / Cout;                    // ks*2 + seg
    int k    = (sk >> 1) * 32 + (sk & 1) * 16 + e;
    Wpk[i] = (k < Cin) ? (bf16)W[(size_t)k * Cout + n] : (bf16)0.0f;
}

// ------------------------- WMMA matmul: OUT = act(IN) @ W -------------------
// Grid: (NN/16, Cout/128), block = 256 (8 waves). Wave w owns a 16x16 tile at
// col = blockIdx.y*128 + w*16. A tile (16x32) staged through LDS as bf16
// pairs with fused act(v) = relu(v*s + t) when ACT. K loop fully unrolled.
template <int CIN, bool ACT>
__global__ __launch_bounds__(256)
void k_mm_wmma(const float* __restrict__ in, const bf16* __restrict__ Wpk,
               const float* __restrict__ s, const float* __restrict__ t,
               int Cout, float* __restrict__ out) {
    constexpr int KSTEPS = (CIN + 31) / 32;
    __shared__ bf16 lds_a[16 * 32];

    const int lane    = threadIdx.x & 31;
    const int wave    = threadIdx.x >> 5;
    const int rowBase = blockIdx.x * 16;
    const int nCol    = blockIdx.y * 128 + wave * 16 + (lane & 15);
    const int m       = lane & 15;
    const int seg     = lane >> 4;          // 0 or 1
    const int half8   = seg * 8;            // C row group

    // staging coords: one bf16 pair per thread (16 rows x 16 pairs)
    const int sr = threadIdx.x >> 4;        // row 0..15
    const int sp = threadIdx.x & 15;        // pair 0..15

    v8f acc = {};
#pragma unroll
    for (int ks = 0; ks < KSTEPS; ++ks) {
        const int kg = ks * 32 + sp * 2;
        float v0 = 0.0f, v1 = 0.0f;
        if (kg + 1 < CIN) {                 // compile-time-resolvable per ks
            const float2 f2 = *(const float2*)&in[(size_t)(rowBase + sr) * CIN + kg];
            v0 = f2.x; v1 = f2.y;
        } else if (kg < CIN) {
            v0 = in[(size_t)(rowBase + sr) * CIN + kg];
        }
        if (ACT) {
            if (kg < CIN)     v0 = fmaxf(v0 * s[kg] + t[kg], 0.0f);
            if (kg + 1 < CIN) v1 = fmaxf(v1 * s[kg + 1] + t[kg + 1], 0.0f);
        }
        __syncthreads();
        v2bf pr; pr[0] = (bf16)v0; pr[1] = (bf16)v1;
        ((v2bf*)lds_a)[sr * 16 + sp] = pr;  // packed ds_store_b32
        __syncthreads();

        // A fragment: two 16B LDS chunks per lane
        // chunks of row m: [k0..7][k8..15][k16..23][k24..31]
        const v8bf* rowp = (const v8bf*)(lds_a + m * 32);
        v8bf alo = rowp[2 * seg];
        v8bf ahi = rowp[2 * seg + 1];
        v16bf af = __builtin_shufflevector(alo, ahi, 0, 1, 2, 3, 4, 5, 6, 7,
                                           8, 9, 10, 11, 12, 13, 14, 15);
        // B fragment: two contiguous 16B global loads from packed weights
        const v8bf* bp = (const v8bf*)(Wpk + ((size_t)(ks * 2 + seg) * Cout + nCol) * 16);
        v8bf blo = bp[0];
        v8bf bhi = bp[1];
        v16bf bfr = __builtin_shufflevector(blo, bhi, 0, 1, 2, 3, 4, 5, 6, 7,
                                            8, 9, 10, 11, 12, 13, 14, 15);

        acc = __builtin_amdgcn_wmma_f32_16x16x32_bf16(false, af, false, bfr,
                                                      (short)0, acc, false, false);
    }
    // C/D 16x16 f32: lane holds col nCol, rows half8+0..7 across the 8 VGPRs
#pragma unroll
    for (int i = 0; i < 8; ++i)
        out[(size_t)(rowBase + half8 + i) * Cout + nCol] = acc[i];
}

// --------------------- atomic-free CSR aggregation --------------------------
// out[i][c] = xw[i][c]*dinv[i]^2 + sum_{k in row i} xw[col[k]][c] * wgt[k]
// One block per node, one thread per channel (coalesced gathers; features in L2).
__global__ void k_agg(const float* __restrict__ xw, const float* __restrict__ dinv,
                      const int* __restrict__ rowptr, const int* __restrict__ col,
                      const float* __restrict__ wgt, int C, float* __restrict__ out) {
    int i = blockIdx.x;
    int c = threadIdx.x;
    float di  = dinv[i];
    float acc = xw[(size_t)i * C + c] * di * di;   // self-loop term
    int beg = rowptr[i], end = rowptr[i + 1];
    for (int k = beg; k < end; ++k) {
        int   sN = col[k];
        float w  = wgt[k];
        acc += xw[(size_t)sN * C + c] * w;
    }
    out[(size_t)i * C + c] = acc;
}

// ------------------------------- pooling -----------------------------------
__global__ void k_pool_acc(const float* __restrict__ h, const int* __restrict__ batch,
                           const float* __restrict__ s3, const float* __restrict__ t3,
                           float* __restrict__ pooled, float* __restrict__ cnt, int C) {
    int i = blockIdx.x;
    int c = threadIdx.x;
    int g = batch[i];
    float v = h[(size_t)i * C + c];
    v = fmaxf(v * s3[c] + t3[c], 0.0f);            // BN3 + ReLU fused
    atomicAdd(&pooled[(size_t)g * C + c], v);
    if (c == 0) atomicAdd(&cnt[g], 1.0f);
}
__global__ void k_pool_div(float* __restrict__ pooled, const float* __restrict__ cnt, int C) {
    int g = blockIdx.x;
    int c = threadIdx.x;
    pooled[(size_t)g * C + c] /= fmaxf(cnt[g], 1.0f);
}

// ----------------------------- small FC head -------------------------------
// out[r][o] = dot(in[r], W[:,o]); if s: relu(dot*s+t) (bias folded in t);
// else dot + bias.
__global__ void k_fc(const float* __restrict__ in, const float* __restrict__ W,
                     int Ci, int Co, const float* __restrict__ s,
                     const float* __restrict__ t, const float* __restrict__ bias,
                     float* __restrict__ out) {
    int r = blockIdx.x;
    int o = threadIdx.x;
    if (o >= Co) return;
    float acc = 0.0f;
    for (int k = 0; k < Ci; ++k)
        acc += in[(size_t)r * Ci + k] * W[(size_t)k * Co + o];
    if (s)   acc = fmaxf(acc * s[o] + t[o], 0.0f);
    if (bias) acc += bias[o];
    out[(size_t)r * Co + o] = acc;
}

// ------------------------------- launcher ----------------------------------
extern "C" void kernel_launch(void* const* d_in, const int* in_sizes, int n_in,
                              void* d_out, int out_size, void* d_ws, size_t ws_size,
                              hipStream_t stream) {
    (void)in_sizes; (void)n_in; (void)out_size; (void)ws_size;
    // inputs (setup_inputs dict order)
    const float* x      = (const float*)d_in[0];
    const int*   esrc   = (const int*)d_in[1];
    const int*   edst   = esrc + NE;
    const int*   batch  = (const int*)d_in[2];
    const float* W1 = (const float*)d_in[3];  const float* b1 = (const float*)d_in[4];
    const float* W2 = (const float*)d_in[5];  const float* b2 = (const float*)d_in[6];
    const float* W3 = (const float*)d_in[7];  const float* b3 = (const float*)d_in[8];
    const float* bn_g[5], *bn_b[5], *bn_m[5], *bn_v[5];
    for (int i = 0; i < 5; ++i) {
        bn_g[i] = (const float*)d_in[9 + 4 * i + 0];
        bn_b[i] = (const float*)d_in[9 + 4 * i + 1];
        bn_m[i] = (const float*)d_in[9 + 4 * i + 2];
        bn_v[i] = (const float*)d_in[9 + 4 * i + 3];
    }
    const float* fc1_W = (const float*)d_in[29]; const float* fc1_b = (const float*)d_in[30];
    const float* fc2_W = (const float*)d_in[31]; const float* fc2_b = (const float*)d_in[32];
    const float* fco_W = (const float*)d_in[33]; const float* fco_b = (const float*)d_in[34];
    float* outp = (float*)d_out;

    // workspace bump allocator (256B aligned)
    char* p = (char*)d_ws;
    auto alloc = [&](size_t bytes) -> void* {
        void* r = (void*)p;
        p += (bytes + 255) & ~(size_t)255;
        return r;
    };
    float* A      = (float*)alloc((size_t)NN * 256 * 4);
    float* B      = (float*)alloc((size_t)NN * 256 * 4);
    float* dinv   = (float*)alloc((size_t)NN * 4);
    int*   cnts   = (int*)  alloc((size_t)NN * 4);
    int*   rowptr = (int*)  alloc((size_t)(NN + 1) * 4);
    int*   cursor = (int*)  alloc((size_t)NN * 4);
    int*   col    = (int*)  alloc((size_t)NE * 4);
    float* wgt    = (float*)alloc((size_t)NE * 4);
    bf16*  wpk    = (bf16*) alloc((size_t)8 * 2 * 256 * 16 * 2);  // max: W2 packed
    float* bnS[5], *bnT[5];
    const int bnC[5] = {256, 256, 128, 128, 64};
    for (int i = 0; i < 5; ++i) { bnS[i] = (float*)alloc(bnC[i] * 4); bnT[i] = (float*)alloc(bnC[i] * 4); }
    float* pooled = (float*)alloc((size_t)NG * 128 * 4);
    float* gcnt   = (float*)alloc((size_t)NG * 4);
    float* m1     = (float*)alloc((size_t)NG * 128 * 4);
    float* m2     = (float*)alloc((size_t)NG * 64 * 4);

    const int TB = 256;
    // --- fold BN params (GCN/FC biases folded into shift) ---
    const float* fold_bias[5] = {b1, b2, b3, fc1_b, fc2_b};
    for (int i = 0; i < 5; ++i)
        k_bn_prep<<<(bnC[i] + TB - 1) / TB, TB, 0, stream>>>(
            bn_g[i], bn_b[i], bn_m[i], bn_v[i], fold_bias[i], bnC[i], bnS[i], bnT[i]);

    // --- degrees / symmetric norm ---
    k_fill_f32<<<(NN + TB - 1) / TB, TB, 0, stream>>>(dinv, 1.0f, NN);   // self-loop
    k_count_deg<<<(NE + TB - 1) / TB, TB, 0, stream>>>(edst, dinv);
    k_rsqrt_inplace<<<(NN + TB - 1) / TB, TB, 0, stream>>>(dinv, NN);

    // --- CSR build (reused by all 3 layers) ---
    k_zero_i32<<<(NN + TB - 1) / TB, TB, 0, stream>>>(cnts, NN);
    k_hist<<<(NE + TB - 1) / TB, TB, 0, stream>>>(edst, cnts);
    k_scan<<<1, 1024, 0, stream>>>(cnts, NN, rowptr);
    k_copy_i32<<<(NN + TB - 1) / TB, TB, 0, stream>>>(cursor, rowptr, NN);
    k_fill_csr<<<(NE + TB - 1) / TB, TB, 0, stream>>>(esrc, edst, dinv, cursor, col, wgt);

    // --- layer 1: x(34) @ W1 -> 256, aggregate ---
    k_pack_w<<<(2 * 2 * 256 * 16 + TB - 1) / TB, TB, 0, stream>>>(W1, 34, 256, 2, wpk);
    k_mm_wmma<34, false><<<dim3(NN / 16, 2), 256, 0, stream>>>(x, wpk, nullptr, nullptr, 256, A);
    k_agg<<<NN, 256, 0, stream>>>(A, dinv, rowptr, col, wgt, 256, B);

    // --- layer 2: relu(bn1(B)) @ W2 -> 256, aggregate ---
    k_pack_w<<<(8 * 2 * 256 * 16 + TB - 1) / TB, TB, 0, stream>>>(W2, 256, 256, 8, wpk);
    k_mm_wmma<256, true><<<dim3(NN / 16, 2), 256, 0, stream>>>(B, wpk, bnS[0], bnT[0], 256, A);
    k_agg<<<NN, 256, 0, stream>>>(A, dinv, rowptr, col, wgt, 256, B);

    // --- layer 3: relu(bn2(B)) @ W3 -> 128, aggregate ---
    k_pack_w<<<(8 * 2 * 128 * 16 + TB - 1) / TB, TB, 0, stream>>>(W3, 256, 128, 8, wpk);
    k_mm_wmma<256, true><<<dim3(NN / 16, 1), 256, 0, stream>>>(B, wpk, bnS[1], bnT[1], 128, A);
    k_agg<<<NN, 128, 0, stream>>>(A, dinv, rowptr, col, wgt, 128, B);

    // --- global mean pool (BN3+ReLU fused into accumulation) ---
    k_fill_f32<<<(NG * 128 + TB - 1) / TB, TB, 0, stream>>>(pooled, 0.0f, NG * 128);
    k_fill_f32<<<(NG + TB - 1) / TB, TB, 0, stream>>>(gcnt, 0.0f, NG);
    k_pool_acc<<<NN, 128, 0, stream>>>(B, batch, bnS[2], bnT[2], pooled, gcnt, 128);
    k_pool_div<<<NG, 128, 0, stream>>>(pooled, gcnt, 128);

    // --- MLP head ---
    k_fc<<<NG, 128, 0, stream>>>(pooled, fc1_W, 128, 128, bnS[3], bnT[3], nullptr, m1);
    k_fc<<<NG, 64, 0, stream>>>(m1, fc2_W, 128, 64, bnS[4], bnT[4], nullptr, m2);
    k_fc<<<NG, 64, 0, stream>>>(m2, fco_W, 64, 4, nullptr, nullptr, fco_b, outp);
}
